// RNNNER_66881230733788
// MI455X (gfx1250) — matrix operand order repaired and checked by reading, not compiled, not run on hardware
//
#include <hip/hip_runtime.h>
#include <hip/hip_bf16.h>

// ---------------------------------------------------------------------------
// LSTM (B=128, S=512, E=128, R=256, C=9) for gfx1250 (CDNA5, wave32).
//
//  * 8 workgroups x 512 threads (16 waves). Each workgroup owns 16 batch rows
//    and runs the full 512-step recurrence privately (no inter-WG sync).
//  * Step GEMM [16, 384(=E+R)] @ [384, 1024] via v_wmma_f32_16x16x32_bf16.
//    Wave w owns gate tiles {w, w+16, w+32, w+48} -> i/f/g/o for hidden cols
//    w*16..w*16+15: gate math fully in-register, c state lives in VGPRs.
//  * W,U pre-packed (bf16, WMMA B-fragment layout) in workspace; L2-resident.
//    B fragments streamed through laundered addrspace(1) pointers (defeats
//    LICM hoist-and-spill, keeps pure-GLOBAL loads) with a DEPTH-2 rotating
//    prefetch pipeline; A fragments (x|h) use a depth-1 LDS pipeline.
//  * Embedding gather software-pipelined one timestep ahead (overlaps GEMM).
//  * h round-trips through LDS (bf16) each step; softmax head fused per step.
// ---------------------------------------------------------------------------

#define GLB __attribute__((address_space(1)))

typedef __attribute__((ext_vector_type(16))) __bf16 v16bf;
typedef __attribute__((ext_vector_type(8)))  float  v8f;
typedef __attribute__((ext_vector_type(4)))  unsigned int u32x4;  // builtin vec:
// unlike HIP's class-based uint4, this loads/stores cleanly from addrspace(1).

union BF16x16 {
    v16bf v;
    u32x4 q[2];
};

#define EMBED   128
#define RNN     256
#define FOURR   1024
#define NCLASS  9
#define BATCH   128
#define SEQ     512
#define NTILES  64         // FOURR/16
#define KT_W    4          // EMBED/32
#define KT_U    8          // RNN/32
#define KT_ALL  (KT_W + KT_U)
#define WP_ELEMS (NTILES * KT_W * 512)   // 131072 bf16
#define UP_ELEMS (NTILES * KT_U * 512)   // 262144 bf16

__device__ __forceinline__ float sigmoid_f(float x) {
    return 1.0f / (1.0f + __expf(-x));
}
__device__ __forceinline__ float tanh_f(float x) {
    x = fminf(fmaxf(x, -15.0f), 15.0f);
    float e = __expf(2.0f * x);
    return (e - 1.0f) / (e + 1.0f);
}

// Pack a [K, 1024] f32 matrix into bf16 WMMA B-fragments:
//   dst[((tile*KT + kt)*32 + lane)*16 + e] = src[(kt*32 + ((lane&16)>>1) + (e&7) + ((e&8)<<1))*1024
//                                                + tile*16 + (lane&15)]
__global__ void lstm_pack_bfrag(const float* __restrict__ src,
                                __bf16* __restrict__ dst, int KT) {
    int p = blockIdx.x * blockDim.x + threadIdx.x;
    int e    = p & 15;
    int lane = (p >> 4) & 31;
    int kt   = (p >> 9) % KT;
    int tile = (p >> 9) / KT;
    int k    = kt * 32 + ((lane & 16) >> 1) + (e & 7) + ((e & 8) << 1);
    int col  = tile * 16 + (lane & 15);
    dst[p] = (__bf16)src[k * FOURR + col];
}

__global__ __launch_bounds__(512, 1)
void lstm_persistent(const int*   __restrict__ ids,     // [B,S]
                     const float* __restrict__ embed,   // [30000,E]
                     const float* __restrict__ bias,    // [4R]
                     const float* __restrict__ wout,    // [R,C]
                     const float* __restrict__ boutp,   // [C]
                     const __bf16* __restrict__ wpack,
                     const __bf16* __restrict__ upack,
                     float* __restrict__ out) {
    __shared__ __align__(16) __bf16 sh_h[16 * RNN];     // 8 KB  (batch-row major)
    __shared__ __align__(16) __bf16 sh_x[16 * EMBED];   // 4 KB
    __shared__ __align__(16) float  sh_wout[RNN * NCLASS];
    __shared__ float sh_bout[16];

    const int tid  = threadIdx.x;
    const int w    = tid >> 5;        // wave 0..15
    const int lane = tid & 31;
    const int l15  = lane & 15;
    const int sel  = lane >> 4;       // 0: lanes 0-15, 1: lanes 16-31
    const int wg   = blockIdx.x;      // 0..7, owns batch rows wg*16..+15

    // --- one-time setup ---
    for (int i = tid; i < 16 * RNN; i += 512) sh_h[i] = (__bf16)0.0f;
    for (int i = tid; i < RNN * NCLASS; i += 512) sh_wout[i] = wout[i];
    if (tid < NCLASS) sh_bout[tid] = boutp[tid];

    float bz[4];
    int   woff[4], uoff[4];   // element offsets of this wave's fragments
#pragma unroll
    for (int nt = 0; nt < 4; ++nt) {
        bz[nt] = bias[nt * RNN + w * 16 + l15];
        const int tile = nt * 16 + w;
        woff[nt] = tile * (KT_W * 512) + lane * 16;
        uoff[nt] = tile * (KT_U * 512) + lane * 16;
    }

    v8f   c_reg = {0.f, 0.f, 0.f, 0.f, 0.f, 0.f, 0.f, 0.f};
    float h_reg[8] = {0.f, 0.f, 0.f, 0.f, 0.f, 0.f, 0.f, 0.f};

    float* out_probs = out;                                   // [B,S,C]
    float* out_h     = out + BATCH * SEQ * NCLASS;            // [B,R]
    float* out_c     = out_h + BATCH * RNN;                   // [B,R]

    // ---- prefetch embedding row for t = 0 (wave w owns batch row w) ----
    const int myrow = wg * 16 + w;
    float4 xv;
    {
        int id = ids[myrow * SEQ + 0];
        xv = *(const float4*)(embed + (long)id * EMBED + lane * 4);
    }

#pragma unroll 1
    for (int t = 0; t < SEQ; ++t) {
        // Opaque, global-address-space copies of the packed-weight pointers.
        // The empty asm is a black box re-evaluated every iteration, so the
        // B-fragment loads cannot be hoisted out of the time loop (which would
        // spill 384 VGPRs of fragments to scratch). The explicit addrspace(1)
        // keeps them on the GLOBAL path (global_load_b128, pure LOADcnt)
        // instead of decaying to FLAT (LOADcnt+DScnt coupling).
        GLB const __bf16* wp_i = (GLB const __bf16*)wpack;
        GLB const __bf16* up_i = (GLB const __bf16*)upack;
        asm volatile("" : "+s"(wp_i), "+s"(up_i));

        // ---- stage 1: store prefetched x row into sh_x ----
        {
            __bf16* dst = sh_x + w * EMBED + lane * 4;
            dst[0] = (__bf16)xv.x; dst[1] = (__bf16)xv.y;
            dst[2] = (__bf16)xv.z; dst[3] = (__bf16)xv.w;
        }
        __syncthreads();  // B1: x visible; prev-step h visible

        // LDS base pointers for A fragments (16-bit A layout: lane<16 holds
        // K {0-7,16-23}; lane>=16 holds K {8-15,24-31} within each 32-K tile).
        const u32x4* px = (const u32x4*)(sh_x + l15 * EMBED);  // 16 vec4/row
        const u32x4* ph = (const u32x4*)(sh_h + l15 * RNN);    // 32 vec4/row

        // ---- prefetch embedding row for t+1 (overlaps the GEMM below) ----
        if (t + 1 < SEQ) {
            int id = ids[myrow * SEQ + t + 1];
            xv = *(const float4*)(embed + (long)id * EMBED + lane * 4);
        }

        // ---- stage 3: z = x@W + h@U + b ----
        // Depth-2 rotating pipeline for B fragments (global/L2), depth-1 for
        // A fragments (LDS): loads for K-step kk+2 (B) / kk+1 (A) issue
        // before the 4 WMMAs of K-step kk.
        v8f z[4];
        BF16x16 a2[2];        // A-fragment rotating buffer
        BF16x16 bb[3][4];     // B-fragment rotating buffer (3 x 4 gate tiles)

#define LOAD_A(j, dst)                                                        \
    do {                                                                      \
        if ((j) < KT_W) {                                                     \
            (dst).q[0] = px[(j) * 4 + sel];                                   \
            (dst).q[1] = px[(j) * 4 + sel + 2];                               \
        } else {                                                              \
            (dst).q[0] = ph[((j) - KT_W) * 4 + sel];                          \
            (dst).q[1] = ph[((j) - KT_W) * 4 + sel + 2];                      \
        }                                                                     \
    } while (0)

#define LOAD_B(kn, buf)                                                       \
    do {                                                                      \
        _Pragma("unroll")                                                     \
        for (int nt = 0; nt < 4; ++nt) {                                      \
            GLB const __bf16* base = ((kn) < KT_W)                            \
                ? (wp_i + woff[nt] + (kn) * 512)                              \
                : (up_i + uoff[nt] + ((kn) - KT_W) * 512);                    \
            GLB const u32x4* p = (GLB const u32x4*)base;                      \
            (buf)[nt].q[0] = p[0];                                            \
            (buf)[nt].q[1] = p[1];                                            \
        }                                                                     \
    } while (0)

#pragma unroll
        for (int nt = 0; nt < 4; ++nt) {
            const float bv = bz[nt];
            z[nt] = (v8f){bv, bv, bv, bv, bv, bv, bv, bv};
        }
        LOAD_B(0, bb[0]);
        LOAD_B(1, bb[1]);
        LOAD_A(0, a2[0]);
        LOAD_A(1, a2[1]);

#pragma unroll
        for (int kk = 0; kk < KT_ALL; ++kk) {
            if (kk + 2 < KT_ALL) LOAD_B(kk + 2, bb[(kk + 2) % 3]);
            if (kk + 1 < KT_ALL) LOAD_A(kk + 1, a2[(kk + 1) & 1]);
            const v16bf a = a2[kk & 1].v;
#pragma unroll
            for (int nt = 0; nt < 4; ++nt) {
                z[nt] = __builtin_amdgcn_wmma_f32_16x16x32_bf16(
                    false, a, false, bb[kk % 3][nt].v, (short)0, z[nt],
                    false, false);
            }
        }
#undef LOAD_A
#undef LOAD_B
        __syncthreads();  // B2: all reads of prev h done before overwrite

        // ---- stage 4: gates, cell update, write new h to LDS ----
        // C-layout: VGPR r, lane -> row M = r + 8*sel, col N = l15
#pragma unroll
        for (int r = 0; r < 8; ++r) {
            float gi = sigmoid_f(z[0][r]);
            float gf = sigmoid_f(z[1][r]);
            float gg = tanh_f(z[2][r]);
            float go = sigmoid_f(z[3][r]);
            float cn = gf * c_reg[r] + gi * gg;
            c_reg[r] = cn;
            float hh = go * tanh_f(cn);
            h_reg[r] = hh;
            sh_h[(r + 8 * sel) * RNN + w * 16 + l15] = (__bf16)hh;
        }
        __syncthreads();  // B3: new h visible

        // ---- stage 5: fused softmax head; wave w handles batch row w ----
        {
            float acc[NCLASS];
#pragma unroll
            for (int cc = 0; cc < NCLASS; ++cc) acc[cc] = 0.0f;
#pragma unroll
            for (int k8 = 0; k8 < 8; ++k8) {
                int k = lane + k8 * 32;
                float hv = (float)sh_h[w * RNN + k];
                const float* wr = sh_wout + k * NCLASS;
#pragma unroll
                for (int cc = 0; cc < NCLASS; ++cc) acc[cc] += hv * wr[cc];
            }
#pragma unroll
            for (int off = 16; off > 0; off >>= 1) {
#pragma unroll
                for (int cc = 0; cc < NCLASS; ++cc)
                    acc[cc] += __shfl_xor(acc[cc], off, 32);
            }
            float lg[NCLASS], m = -3.0e38f;
#pragma unroll
            for (int cc = 0; cc < NCLASS; ++cc) {
                lg[cc] = acc[cc] + sh_bout[cc];
                m = fmaxf(m, lg[cc]);
            }
            float s = 0.0f;
#pragma unroll
            for (int cc = 0; cc < NCLASS; ++cc) {
                float e = __expf(lg[cc] - m);
                lg[cc] = e;
                s += e;
            }
            float inv = 1.0f / s;
            if (lane == 0) {
                float* po = out_probs + ((long)myrow * SEQ + t) * NCLASS;
#pragma unroll
                for (int cc = 0; cc < NCLASS; ++cc) po[cc] = lg[cc] * inv;
            }
        }
    }

    // ---- final states ----
#pragma unroll
    for (int r = 0; r < 8; ++r) {
        int gb  = wg * 16 + r + 8 * sel;
        int col = w * 16 + l15;
        out_h[gb * RNN + col] = h_reg[r];
        out_c[gb * RNN + col] = c_reg[r];
    }
}

extern "C" void kernel_launch(void* const* d_in, const int* in_sizes, int n_in,
                              void* d_out, int out_size, void* d_ws, size_t ws_size,
                              hipStream_t stream) {
    const int*   ids   = (const int*)d_in[0];     // [128,512]
    const float* embed = (const float*)d_in[1];   // [30000,128]
    const float* W     = (const float*)d_in[2];   // [128,1024]
    const float* U     = (const float*)d_in[3];   // [256,1024]
    const float* b     = (const float*)d_in[4];   // [1024]
    const float* Wout  = (const float*)d_in[5];   // [256,9]
    const float* bout  = (const float*)d_in[6];   // [9]
    float* out = (float*)d_out;

    __bf16* wp = (__bf16*)d_ws;
    __bf16* up = wp + WP_ELEMS;

    lstm_pack_bfrag<<<WP_ELEMS / 256, 256, 0, stream>>>(W, wp, KT_W);
    lstm_pack_bfrag<<<UP_ELEMS / 256, 256, 0, stream>>>(U, up, KT_U);

    lstm_persistent<<<BATCH / 16, 512, 0, stream>>>(
        ids, embed, b, Wout, bout, wp, up, out);
}
